// DiagBlockAttention_23948737643217
// MI455X (gfx1250) — matrix operand order repaired and checked by reading, not compiled, not run on hardware
//
#include <hip/hip_runtime.h>
#include <hip/hip_bf16.h>
#include <stdint.h>
#include <stddef.h>

// ---------------------------------------------------------------------------
// CDNA5 (gfx1250) wave32 WMMA implementation of DiagBlockAttention.
//   B=16, N=3136 (=56x56), DIM=768, H=12, DH=64, block = 4x4 patch = 16 tokens
// GEMMs use async-to-LDS (ASYNCcnt) double-buffered tile staging + WMMA bf16.
// ---------------------------------------------------------------------------

typedef __attribute__((ext_vector_type(16))) __bf16       v16bf;
typedef __attribute__((ext_vector_type(8)))  float        v8f;
typedef __attribute__((ext_vector_type(4)))  unsigned int u32x4;

#define WMMA_BF16(a, b, c) \
  __builtin_amdgcn_wmma_f32_16x16x32_bf16(false, (a), false, (b), (short)0, (c), false, false)

__device__ __forceinline__ unsigned short f2bf(float f) {
  unsigned int u = __float_as_uint(f);
  u += 0x7FFFu + ((u >> 16) & 1u);   // round-to-nearest-even
  return (unsigned short)(u >> 16);
}

union FragBF {                 // one 16x32 bf16 A-frag or 32x16 B-frag (8 VGPRs/lane)
  u32x4          u4[2];
  unsigned int   w[8];
  unsigned short s[16];
  v16bf          v;
};

// generic pointer into __shared__ -> LDS byte offset (low 32 bits of generic addr)
__device__ __forceinline__ unsigned ldsoff(const void* p) {
  return (unsigned)(uintptr_t)p;
}

// per-lane async DMA: LDS[dst] <- 16 bytes at global src (tracked by ASYNCcnt)
__device__ __forceinline__ void async_ld_b128(unsigned lds_dst, const void* gsrc) {
  asm volatile("global_load_async_to_lds_b128 %0, %1, off"
               :: "v"(lds_dst), "v"(gsrc) : "memory");
}

__device__ __forceinline__ void wait_async0() {
#if __has_builtin(__builtin_amdgcn_s_wait_asynccnt)
  __builtin_amdgcn_s_wait_asynccnt(0);
#else
  asm volatile("s_wait_asynccnt 0x0" ::: "memory");
#endif
}

// ------------------------- conversion kernels ------------------------------

__global__ __launch_bounds__(256) void cvt_f32_bf16(const float* __restrict__ src,
                                                    unsigned short* __restrict__ dst,
                                                    size_t n) {
  size_t i = (size_t)blockIdx.x * blockDim.x + threadIdx.x;
  if (i < n) dst[i] = f2bf(src[i]);
}

// w[K][N] (row-major) -> wt[N][K]  (K-contiguous per output column)
__global__ __launch_bounds__(256) void cvt_transpose_bf16(const float* __restrict__ w,
                                                          unsigned short* __restrict__ wt,
                                                          int K, int N) {
  int i = blockIdx.x * blockDim.x + threadIdx.x;
  if (i < K * N) {
    int k = i / N, n = i % N;
    wt[(size_t)n * K + k] = f2bf(w[i]);
  }
}

// ------------------------------ WMMA GEMM ----------------------------------
// C[M,N] = A[M,K](bf16 row-major) x Bt[N,K](bf16, K contiguous per col)
// 256 threads = 8 waves; tile 128x128; wave -> 64x32 subtile (8 accumulators).
// K-step 32. Tiles staged in LDS via async-to-LDS DMA, double buffered so the
// next tile's DMA overlaps the current tile's WMMAs.
// MODE 0: store bf16.  MODE 1: add f32 bias, store f32.

template <int MODE>
__global__ __launch_bounds__(256) void gemm_bf16_wmma(
    const unsigned short* __restrict__ A,
    const unsigned short* __restrict__ Bt,
    unsigned short* __restrict__ Cb,
    float* __restrict__ Cf,
    const float* __restrict__ bias,
    int M, int N, int K) {
  __shared__ unsigned short sA[2][128 * 32];   // [buf][row][32 K-halves] 8KB each
  __shared__ unsigned short sB[2][128 * 32];   // [buf][col][32 K-halves]

  const int tid  = threadIdx.x;
  const int lane = tid & 31;
  const int wave = tid >> 5;
  const int l16  = lane & 15;
  const int hi   = lane >> 4;          // which half of the wave
  const int wr   = wave >> 2;          // 0..1 -> 64 rows each
  const int wc   = wave & 3;           // 0..3 -> 32 cols each
  const int rowBase = blockIdx.x * 128 + wr * 64;
  const int colBase = blockIdx.y * 128 + wc * 32;

  const unsigned short* gA = A  + (size_t)(blockIdx.x * 128) * K;
  const unsigned short* gB = Bt + (size_t)(blockIdx.y * 128) * K;

  // tile = 128 rows x 32 halves = 512 b128-chunks; chunk c -> row c>>2,
  // K-halves (c&3)*8, LDS halves offset c*8. Each thread moves 2 A + 2 B chunks.
  auto issueTile = [&](int buf, int k0) {
#pragma unroll
    for (int q = 0; q < 2; ++q) {
      const int c   = tid + q * 256;
      const int row = c >> 2;
      const int col = (c & 3) * 8;
      async_ld_b128(ldsoff(&sA[buf][c * 8]), gA + (size_t)row * K + k0 + col);
      async_ld_b128(ldsoff(&sB[buf][c * 8]), gB + (size_t)row * K + k0 + col);
    }
  };

  v8f acc[4][2] = {};

  issueTile(0, 0);
  int buf = 0;
  for (int k0 = 0; k0 < K; k0 += 32) {
    wait_async0();          // my DMA for sX[buf] has landed
    __syncthreads();        // everyone's has; previous reads of sX[buf^1] done
    if (k0 + 32 < K) issueTile(buf ^ 1, k0 + 32);   // overlap DMA with WMMAs

    FragBF af[4], bfr[2];
    // A frag (16x32): lane l16 = row; K chunks [hi*8,+8) and [hi*8+16,+8)
#pragma unroll
    for (int i = 0; i < 4; ++i) {
      const unsigned short* p = &sA[buf][(wr * 64 + i * 16 + l16) * 32 + hi * 8];
      af[i].u4[0] = *(const u32x4*)p;
      af[i].u4[1] = *(const u32x4*)(p + 16);
    }
    // B frag (32x16): lane l16 = col; K chunks [hi*8,+8) and [hi*8+16,+8)
#pragma unroll
    for (int j = 0; j < 2; ++j) {
      const unsigned short* p = &sB[buf][(wc * 32 + j * 16 + l16) * 32 + hi * 8];
      bfr[j].u4[0] = *(const u32x4*)p;
      bfr[j].u4[1] = *(const u32x4*)(p + 16);
    }
#pragma unroll
    for (int i = 0; i < 4; ++i)
#pragma unroll
      for (int j = 0; j < 2; ++j)
        acc[i][j] = WMMA_BF16(af[i].v, bfr[j].v, acc[i][j]);

    __syncthreads();        // reads of sX[buf] done before it is refilled
    buf ^= 1;
  }

  // C/D layout: VGPR r -> row M = r + 8*hi, col N = l16
#pragma unroll
  for (int i = 0; i < 4; ++i) {
    const int row0 = rowBase + i * 16 + hi * 8;
#pragma unroll
    for (int j = 0; j < 2; ++j) {
      const int col = colBase + j * 16 + l16;
      const float bv = (MODE == 1) ? bias[col] : 0.0f;
#pragma unroll
      for (int r = 0; r < 8; ++r) {
        const size_t idx = (size_t)(row0 + r) * N + col;
        if (MODE == 0) Cb[idx] = f2bf(acc[i][j][r]);
        else           Cf[idx] = acc[i][j][r] + bv;
      }
    }
  }
}

// --------------------------- block attention -------------------------------
// One wave handles one (b, h, nb, mb) 16-token block entirely in registers+LDS.
// qkv layout: [B*3136][2304] bf16 (q: +0, k: +768, v: +1536; head h: +h*64)
// o   layout: [B*3136][768]  bf16

__global__ __launch_bounds__(256) void attn_block_wmma(
    const unsigned short* __restrict__ qkv,
    unsigned short* __restrict__ o) {
  __shared__ unsigned short smem[8 * 1280];   // per wave: 16x64 V + 16x16 P

  const int lane = threadIdx.x & 31;
  const int wave = threadIdx.x >> 5;
  const int l16  = lane & 15;
  const int hi   = lane >> 4;

  int g = blockIdx.x * 8 + wave;              // 37632 blocks total, grid=4704
  const int mb = g % 14; g /= 14;
  const int nb = g % 14; g /= 14;
  const int h  = g % 12; g /= 12;
  const int bb = g;

  // token index of block element e (0..15)
  auto tokOf = [&](int e) { return (nb * 4 + (e >> 2)) * 56 + mb * 4 + (e & 3); };
  const int myTok = tokOf(l16);
  const unsigned short* rowp = qkv + ((size_t)bb * 3136 + myTok) * 2304 + h * 64;

  unsigned short* vlds = smem + wave * 1280;          // [16][64] bf16
  unsigned short* plds = smem + wave * 1280 + 1024;   // [16][16] bf16

  // stage V rows into LDS: lane(l16,hi) moves 32 halves of token row l16
  {
    const u32x4* src = (const u32x4*)(rowp + 1536 + hi * 32);
    u32x4* dst = (u32x4*)(vlds + l16 * 64 + hi * 32);
    dst[0] = src[0]; dst[1] = src[1]; dst[2] = src[2]; dst[3] = src[3];
  }

  // S = Q K^T : two chained 16x16x32 bf16 WMMAs over DH=64
  v8f s = {};
#pragma unroll
  for (int t = 0; t < 2; ++t) {
    FragBF qa, kb;
    qa.u4[0] = *(const u32x4*)(rowp + t * 32 + hi * 8);
    qa.u4[1] = *(const u32x4*)(rowp + t * 32 + hi * 8 + 16);
    kb.u4[0] = *(const u32x4*)(rowp + 768 + t * 32 + hi * 8);
    kb.u4[1] = *(const u32x4*)(rowp + 768 + t * 32 + hi * 8 + 16);
    s = WMMA_BF16(qa.v, kb.v, s);
  }

  // softmax: row M = r + 8*hi lives across the 16 lanes of this half-wave
  float p[8];
#pragma unroll
  for (int r = 0; r < 8; ++r) {
    float sv = s[r] * 0.125f;                 // DH^-0.5
    float mx = sv;
#pragma unroll
    for (int msk = 1; msk < 16; msk <<= 1)
      mx = fmaxf(mx, __shfl_xor(mx, msk, 32));
    float e = __expf(sv - mx);
    float sum = e;
#pragma unroll
    for (int msk = 1; msk < 16; msk <<= 1)
      sum += __shfl_xor(sum, msk, 32);
    p[r] = e / sum;
  }

  // C-layout -> A-layout via LDS (16x16 bf16), K padded 16->32 with zeros
#pragma unroll
  for (int r = 0; r < 8; ++r)
    plds[(r + hi * 8) * 16 + l16] = f2bf(p[r]);
  __syncthreads();

  FragBF pa;
  pa.u4[0] = hi ? (u32x4){0u, 0u, 0u, 0u} : *(const u32x4*)(plds + l16 * 16);
  pa.u4[1] = hi ? (u32x4){0u, 0u, 0u, 0u} : *(const u32x4*)(plds + l16 * 16 + 8);

  // O = P V : 4 WMMAs over the 64 dh columns
#pragma unroll
  for (int t = 0; t < 4; ++t) {
    FragBF vb;
    if (hi == 0) {                            // K = tokens 0..15 (real)
#pragma unroll
      for (int v = 0; v < 8; ++v) {
        unsigned int lo = vlds[(2 * v + 0) * 64 + t * 16 + l16];
        unsigned int hh = vlds[(2 * v + 1) * 64 + t * 16 + l16];
        vb.w[v] = lo | (hh << 16);
      }
    } else {                                  // K = 16..31 (zero pad)
      vb.u4[0] = (u32x4){0u, 0u, 0u, 0u};
      vb.u4[1] = (u32x4){0u, 0u, 0u, 0u};
    }
    v8f of = {};
    of = WMMA_BF16(pa.v, vb.v, of);
#pragma unroll
    for (int r = 0; r < 8; ++r) {
      const int tok = tokOf(r + hi * 8);
      o[((size_t)bb * 3136 + tok) * 768 + h * 64 + t * 16 + l16] = f2bf(of[r]);
    }
  }
}

// ------------------------------- launcher ----------------------------------

extern "C" void kernel_launch(void* const* d_in, const int* in_sizes, int n_in,
                              void* d_out, int out_size, void* d_ws, size_t ws_size,
                              hipStream_t stream) {
  const float* x     = (const float*)d_in[0];   // [16,3136,768]
  const float* w_qkv = (const float*)d_in[1];   // [768,2304]
  const float* w_out = (const float*)d_in[2];   // [768,768]
  const float* b_out = (const float*)d_in[3];   // [768]
  float* out = (float*)d_out;                   // [16,3136,768] f32

  // workspace layout (bytes); total = 390,070,272
  char* ws = (char*)d_ws;
  unsigned short* xb    = (unsigned short*)(ws + 0);           //  77,070,336
  unsigned short* wqkvT = (unsigned short*)(ws + 77070336);    //   3,538,944
  unsigned short* woutT = (unsigned short*)(ws + 80609280);    //   1,179,648
  unsigned short* qkvb  = (unsigned short*)(ws + 81788928);    // 231,211,008
  unsigned short* ob    = (unsigned short*)(ws + 312999936);   //  77,070,336

  const size_t nX = (size_t)16 * 3136 * 768;   // 38,535,168
  cvt_f32_bf16<<<(unsigned)((nX + 255) / 256), 256, 0, stream>>>(x, xb, nX);
  cvt_transpose_bf16<<<(768 * 2304 + 255) / 256, 256, 0, stream>>>(w_qkv, wqkvT, 768, 2304);
  cvt_transpose_bf16<<<(768 * 768 + 255) / 256, 256, 0, stream>>>(w_out, woutT, 768, 768);

  // qkv = x @ w_qkv    (M=50176, N=2304, K=768)
  dim3 g1(392, 18);
  gemm_bf16_wmma<0><<<g1, 256, 0, stream>>>(xb, wqkvT, qkvb, nullptr, nullptr,
                                            50176, 2304, 768);

  // block-diagonal attention: 16*12*14*14 = 37632 waves, 8 waves/block
  attn_block_wmma<<<4704, 256, 0, stream>>>(qkvb, ob);

  // out = o @ w_out + b_out   (M=50176, N=768, K=768)
  dim3 g3(392, 6);
  gemm_bf16_wmma<1><<<g3, 256, 0, stream>>>(ob, woutT, nullptr, out, b_out,
                                            50176, 768, 768);
}
